// Layer_68212670595884
// MI455X (gfx1250) — matrix operand order repaired and checked by reading, not compiled, not run on hardware
//
#include <hip/hip_runtime.h>
#include <hip/hip_bf16.h>

#define DIMC   512
#define HEADS  8
#define DHEAD  64
#define SEQ    2048
#define BATCH  4
#define ROWS   (BATCH * SEQ)      // 8192
#define QKVW   (3 * DIMC)         // 1536

typedef __attribute__((ext_vector_type(16))) __bf16 v16bf;
typedef __attribute__((ext_vector_type(8)))  __bf16 v8bf;
typedef __attribute__((ext_vector_type(8)))  float  v8f;
typedef unsigned int u32x4 __attribute__((ext_vector_type(4)));
typedef int          i32x4 __attribute__((ext_vector_type(4)));
typedef int          i32x8 __attribute__((ext_vector_type(8)));

union V16 { v16bf v; v8bf h[2]; };

#define BM 128
#define BN 128
#define BK 32
#define BPAD 40   // LDS K-stride for B tile: 80B rows (16B aligned, conflict-free)

#define FLAG_BETA1   1
#define FLAG_BF16OUT 2
#define FLAG_BIAS    4
#define FLAG_CTRANS  8

// Tensor Data Mover availability (device pass only; host parses fallback path)
#if defined(__HIP_DEVICE_COMPILE__) && defined(__has_builtin)
#  if __has_builtin(__builtin_amdgcn_tensor_load_to_lds) && __has_builtin(__builtin_amdgcn_s_wait_tensorcnt)
#    define USE_TDM 1
#  endif
#endif
#ifndef USE_TDM
#  define USE_TDM 0
#endif

#if USE_TDM
// Issue a TDM load of a [128 rows x 32 elem] bf16 tile (row stride = ld elems)
// into LDS at lds_addr, padding each 64B row with 16B (-> BPAD=40 layout).
// Zero-fills rows beyond rem_n (N tail) per D# OOB rules.
__device__ __forceinline__ void tdm_load_tile_b(const __bf16* gptr, unsigned lds_addr,
                                                int rem_k, int rem_n, int ld)
{
    unsigned long long ga = (unsigned long long)gptr;
    u32x4 g0;
    g0.x = 1u;                                            // count=1, user mode
    g0.y = lds_addr;                                      // LDS byte address
    g0.z = (unsigned)(ga & 0xFFFFFFFFull);                // global_addr[31:0]
    g0.w = (unsigned)((ga >> 32) & 0x1FFFFFFull) | (2u << 30); // addr[56:32] | type=2
    i32x8 g1;
    // data_size=1(2B)@16 | pad_enable@20 | pad_interval=3(16DW)@22 | pad_amount=3(4DW)@25
    g1[0] = (1 << 16) | (1 << 20) | (3 << 22) | (3 << 25);
    g1[1] = (rem_k & 0xFFFF) << 16;                       // tensor_dim0[15:0] @ bits63:48
    g1[2] = ((rem_k >> 16) & 0xFFFF) | ((rem_n & 0xFFFF) << 16); // dim0 hi | dim1 lo
    g1[3] = ((rem_n >> 16) & 0xFFFF) | (BK << 16);        // dim1 hi | tile_dim0=32
    g1[4] = BN;                                           // tile_dim1=128, tile_dim2=0
    g1[5] = ld;                                           // tensor_dim0_stride[31:0]
    g1[6] = 0;                                            // stride hi | dim1_stride lo
    g1[7] = 0;
    i32x4 z4; z4[0] = z4[1] = z4[2] = z4[3] = 0;
#if __clang_major__ >= 23
    i32x8 z8; for (int e = 0; e < 8; ++e) z8[e] = 0;
    __builtin_amdgcn_tensor_load_to_lds(g0, g1, z4, z4, z8, 0);
#else
    __builtin_amdgcn_tensor_load_to_lds(g0, g1, z4, z4, 0);
#endif
}
#endif

// ---------------------------------------------------------------------------
// Generic bf16 WMMA GEMM: C[M,N] (+)= A[M,K] * BT[N,K]^T (+ bias)
// B is supplied TRANSPOSED (row-major N x K, row stride ldb) so both tiles
// stage with vector loads; B tile goes through the Tensor Data Mover.
// Batched via blockIdx.z; A rows may be shifted with per-batch zero padding.
// ---------------------------------------------------------------------------
__global__ __launch_bounds__(256)
void gemm_bf16_wmma(const __bf16* __restrict__ A, const __bf16* __restrict__ BT,
                    const float* __restrict__ bias,
                    float* __restrict__ Cf, __bf16* __restrict__ Cb,
                    int M, int N, int K, int lda, int ldb, int ldc,
                    long long a_bs, long long b_bs,
                    int hdiv, long long c_outer, long long c_inner,
                    int row_shift, int rpb, int flags)
{
    __shared__ __align__(16) __bf16 As[BM][BK];
    __shared__ __align__(16) __bf16 Bs[BN][BPAD];

    const int tid  = threadIdx.x;
    const int lane = tid & 31;
    const int wid  = tid >> 5;
    const int wm   = (wid >> 1) * 32;   // wave row offset: 0,32,64,96
    const int wn   = (wid & 1) * 64;    // wave col offset: 0,64
    const int lhi  = lane >> 4;         // 0/1 (K half select)
    const int llo  = lane & 15;         // row/col within 16

    const int z = blockIdx.z;
    const long long a_off = (long long)z * a_bs;
    const long long b_off = (long long)z * b_bs;
    const long long c_off = (long long)(z / hdiv) * c_outer
                          + (long long)(z % hdiv) * c_inner;

    const int m0 = blockIdx.y * BM;
    const int n0 = blockIdx.x * BN;

    v8f acc[2][4];
    for (int i = 0; i < 2; ++i)
        for (int j = 0; j < 4; ++j)
            for (int e = 0; e < 8; ++e) acc[i][j][e] = 0.0f;

    // tile staging map: thread -> (row, 16-elem K chunk)
    const int sr = tid >> 1;             // 0..127
    const int sc = (tid & 1) * 16;       // 0 or 16

#if USE_TDM
    const unsigned bs_lds = (unsigned)(unsigned long long)&Bs[0][0];
#endif

    for (int kk = 0; kk < K; kk += BK) {
#if USE_TDM
        // ---- stage B via Tensor Data Mover (zero-fills N tail via D# dims)
        if (tid == 0)
            tdm_load_tile_b(BT + b_off + (long long)n0 * ldb + kk,
                            bs_lds, K - kk, N - n0, ldb);
#else
        // ---- stage B synchronously (vectorized, BT row-major N x K)
        {
            int gn = n0 + sr;
            v8bf y0, y1;
            if (gn < N) {
                const v8bf* gp = (const v8bf*)(BT + b_off + (long long)gn * ldb + kk + sc);
                y0 = gp[0]; y1 = gp[1];
            } else {
                for (int e = 0; e < 8; ++e) { y0[e] = (__bf16)0.0f; y1[e] = (__bf16)0.0f; }
            }
            *(v8bf*)&Bs[sr][sc]     = y0;
            *(v8bf*)&Bs[sr][sc + 8] = y1;
        }
#endif
        // ---- stage A (row-major, row-shifted with batch-boundary padding)
        {
            int mlog = m0 + sr;
            int msrc = mlog + row_shift;
            bool ok = (mlog < M) && (msrc >= 0) && (msrc < M);
            if (ok && rpb > 0) ok = (msrc / rpb) == (mlog / rpb);
            v8bf x0, x1;
            if (ok) {
                const __bf16* ap = A + a_off + (long long)msrc * lda + kk + sc;
                const v8bf* gp = (const v8bf*)ap;
                x0 = gp[0]; x1 = gp[1];
                if (kk + BK < K)
                    __builtin_prefetch((const void*)(ap + BK), 0, 0);
            } else {
                for (int e = 0; e < 8; ++e) { x0[e] = (__bf16)0.0f; x1[e] = (__bf16)0.0f; }
            }
            *(v8bf*)&As[sr][sc]     = x0;
            *(v8bf*)&As[sr][sc + 8] = x1;
        }
#if USE_TDM
        if (wid == 0) __builtin_amdgcn_s_wait_tensorcnt(0);
#endif
        __syncthreads();

        // ---- fragments per ISA 16-bit layouts
        V16 afrag[2], bfrag[4];
        #pragma unroll
        for (int i = 0; i < 2; ++i) {
            const __bf16* arow = &As[wm + 16 * i + llo][0];
            afrag[i].h[0] = *(const v8bf*)(arow + lhi * 8);        // K = lhi*8 .. +7
            afrag[i].h[1] = *(const v8bf*)(arow + lhi * 8 + 16);   // K = 16+lhi*8 ..
        }
        #pragma unroll
        for (int j = 0; j < 4; ++j) {
            const __bf16* brow = &Bs[wn + 16 * j + llo][lhi * 8]; // col llo, K = lhi*8.. and 16+lhi*8..
            bfrag[j].h[0] = *(const v8bf*)(brow);
            bfrag[j].h[1] = *(const v8bf*)(brow + 16);
        }
        #pragma unroll
        for (int i = 0; i < 2; ++i)
            #pragma unroll
            for (int j = 0; j < 4; ++j)
                acc[i][j] = __builtin_amdgcn_wmma_f32_16x16x32_bf16(
                    false, afrag[i].v, false, bfrag[j].v,
                    (short)0, acc[i][j], false, false);
        __syncthreads();
    }

    const bool beta1 = flags & FLAG_BETA1;
    const bool asbf  = flags & FLAG_BF16OUT;
    const bool hasb  = flags & FLAG_BIAS;
    const bool ctr   = flags & FLAG_CTRANS;
    #pragma unroll
    for (int i = 0; i < 2; ++i) {
        #pragma unroll
        for (int j = 0; j < 4; ++j) {
            int col = n0 + wn + 16 * j + llo;
            if (col >= N) continue;
            float bv = hasb ? bias[col] : 0.0f;
            #pragma unroll
            for (int r = 0; r < 8; ++r) {
                int m = m0 + wm + 16 * i + r + 8 * lhi;
                if (m >= M) continue;
                long long idx = ctr ? (c_off + (long long)col * ldc + m)
                                    : (c_off + (long long)m * ldc + col);
                float v = acc[i][j][r] + bv;
                if (asbf) {
                    Cb[idx] = (__bf16)v;
                } else {
                    if (beta1) v += Cf[idx];
                    Cf[idx] = v;
                }
            }
        }
    }
}

// ---------------------------------------------------------------------------
// Elementwise / reduction helper kernels (all f32 math)
// ---------------------------------------------------------------------------
__global__ void cvt_f32_bf16(const float* __restrict__ s, __bf16* __restrict__ d, long long n)
{
    long long i = (long long)blockIdx.x * 256 + threadIdx.x;
    if (i < n) d[i] = (__bf16)s[i];
}

// src R x C (row-major f32) -> dst C x R (row-major bf16) : dst = src^T
__global__ void cvt_transpose_f32_bf16(const float* __restrict__ s, __bf16* __restrict__ d,
                                       int R, int C)
{
    long long i = (long long)blockIdx.x * 256 + threadIdx.x;
    if (i >= (long long)R * C) return;
    int r = (int)(i / C), c = (int)(i % C);
    d[(long long)c * R + r] = (__bf16)s[i];
}

// w[o][i][t] (O,I,9) -> wt[t][o][i]  (per-tap B^T = [out_ch][in_ch])
__global__ void conv_w_transpose(const float* __restrict__ w, __bf16* __restrict__ wt, long long n)
{
    long long idx = (long long)blockIdx.x * 256 + threadIdx.x;
    if (idx >= n) return;
    int t = (int)(idx % 9);
    long long oi = idx / 9;
    int ii = (int)(oi % DIMC);
    int o  = (int)(oi / DIMC);
    wt[((long long)t * DIMC + o) * DIMC + ii] = (__bf16)w[idx];
}

__device__ __forceinline__ float rope_val(int n, int d)
{
    int j = d & 31;
    float invf = __expf(-(float)j * 0.03125f * 9.210340371976184f); // 10000^(-j/32)
    float ang = (float)n * invf;
    return (d < 32) ? __sinf(ang) : __cosf(ang);
}

__device__ __forceinline__ float gelu_exact(float x)
{
    return 0.5f * x * (1.0f + erff(x * 0.70710678118654752f));
}

// q: +rope, softmax over d (64), *64^-0.5 -> qp[b,h,n,d]
// v: masked -> vpT[b,h,e,n]   (transposed: B^T operand of the context GEMM)
__global__ __launch_bounds__(64)
void attn_pre(const float* __restrict__ qkv, const unsigned char* __restrict__ mask,
              __bf16* __restrict__ qp, __bf16* __restrict__ vpT)
{
    int d = threadIdx.x;
    int n = blockIdx.x;
    int h = blockIdx.y;
    int b = blockIdx.z;
    long long row = (long long)b * SEQ + n;
    const float* base = qkv + row * QKVW;
    float qv = base[h * DHEAD + d] + rope_val(n, d);

    __shared__ float red[64];
    red[d] = qv; __syncthreads();
    for (int s = 32; s > 0; s >>= 1) {
        if (d < s) red[d] = fmaxf(red[d], red[d + s]);
        __syncthreads();
    }
    float mx = red[0]; __syncthreads();
    float e = __expf(qv - mx);
    red[d] = e; __syncthreads();
    for (int s = 32; s > 0; s >>= 1) {
        if (d < s) red[d] += red[d + s];
        __syncthreads();
    }
    float sum = red[0];

    long long z = (long long)(b * HEADS + h);
    qp[(z * SEQ + n) * DHEAD + d] = (__bf16)(e / sum * 0.125f);   // * DHEAD^-0.5
    float vv = mask[row] ? base[2 * DIMC + h * DHEAD + d] : 0.0f;
    vpT[(z * DHEAD + d) * SEQ + n] = (__bf16)vv;
}

// k: +rope, mask->-big, softmax over n (2048) -> kT[b,h,d,n] (A of context GEMM)
__global__ __launch_bounds__(256)
void k_softmax(const float* __restrict__ qkv, const unsigned char* __restrict__ mask,
               __bf16* __restrict__ kT)
{
    int d = blockIdx.x;
    int h = blockIdx.y;
    int b = blockIdx.z;
    int tid = threadIdx.x;
    const float* base = qkv + (long long)b * SEQ * QKVW + DIMC + h * DHEAD + d;
    const unsigned char* mrow = mask + (long long)b * SEQ;
    const float NEG = -3.402823466e38f;

    __shared__ float red[256];
    float mx = NEG;
    for (int n = tid; n < SEQ; n += 256) {
        float x = mrow[n] ? (base[(long long)n * QKVW] + rope_val(n, d)) : NEG;
        mx = fmaxf(mx, x);
    }
    red[tid] = mx; __syncthreads();
    for (int s = 128; s > 0; s >>= 1) {
        if (tid < s) red[tid] = fmaxf(red[tid], red[tid + s]);
        __syncthreads();
    }
    mx = red[0]; __syncthreads();

    float sum = 0.0f;
    for (int n = tid; n < SEQ; n += 256) {
        float x = mrow[n] ? (base[(long long)n * QKVW] + rope_val(n, d)) : NEG;
        sum += __expf(x - mx);
    }
    red[tid] = sum; __syncthreads();
    for (int s = 128; s > 0; s >>= 1) {
        if (tid < s) red[tid] += red[tid + s];
        __syncthreads();
    }
    float inv = 1.0f / red[0];

    long long zz = (long long)(b * HEADS + h);
    __bf16* out = kT + (zz * DHEAD + d) * (long long)SEQ;
    for (int n = tid; n < SEQ; n += 256) {
        float x = mrow[n] ? (base[(long long)n * QKVW] + rope_val(n, d)) : NEG;
        out[n] = (__bf16)(__expf(x - mx) * inv);
    }
}

__global__ void make_conv_in(const float* __restrict__ tok,
                             const unsigned char* __restrict__ mask,
                             __bf16* __restrict__ xm)
{
    long long i = (long long)blockIdx.x * 256 + threadIdx.x;
    if (i >= (long long)ROWS * DIMC) return;
    long long row = i >> 9;   // / DIMC
    xm[i] = mask[row] ? (__bf16)tok[i] : (__bf16)0.0f;
}

// tokens + gelu(narrow) + gelu(wide), then layernorm(ln1) -> bf16 for ff GEMM
__global__ __launch_bounds__(256)
void combine_ln1(const float* __restrict__ tok, const float* __restrict__ nacc,
                 const float* __restrict__ wacc, const float* __restrict__ g,
                 const float* __restrict__ bb, __bf16* __restrict__ out)
{
    long long base = (long long)blockIdx.x * DIMC;
    int tid = threadIdx.x;
    int c0 = tid, c1 = tid + 256;
    __shared__ float rs[256], rq[256];
    float v0 = tok[base + c0] + gelu_exact(nacc[base + c0]) + gelu_exact(wacc[base + c0]);
    float v1 = tok[base + c1] + gelu_exact(nacc[base + c1]) + gelu_exact(wacc[base + c1]);
    rs[tid] = v0 + v1; rq[tid] = v0 * v0 + v1 * v1; __syncthreads();
    for (int s = 128; s > 0; s >>= 1) {
        if (tid < s) { rs[tid] += rs[tid + s]; rq[tid] += rq[tid + s]; }
        __syncthreads();
    }
    float mean = rs[0] * (1.0f / DIMC);
    float var  = rq[0] * (1.0f / DIMC) - mean * mean;
    float inv  = rsqrtf(var + 1e-5f);
    out[base + c0] = (__bf16)((v0 - mean) * inv * g[c0] + bb[c0]);
    out[base + c1] = (__bf16)((v1 - mean) * inv * g[c1] + bb[c1]);
}

// layernorm(gelu(ff_out)) -> final f32 output
__global__ __launch_bounds__(256)
void gelu_ln2(const float* __restrict__ ff, const float* __restrict__ g,
              const float* __restrict__ bb, float* __restrict__ out)
{
    long long base = (long long)blockIdx.x * DIMC;
    int tid = threadIdx.x;
    int c0 = tid, c1 = tid + 256;
    __shared__ float rs[256], rq[256];
    float v0 = gelu_exact(ff[base + c0]);
    float v1 = gelu_exact(ff[base + c1]);
    rs[tid] = v0 + v1; rq[tid] = v0 * v0 + v1 * v1; __syncthreads();
    for (int s = 128; s > 0; s >>= 1) {
        if (tid < s) { rs[tid] += rs[tid + s]; rq[tid] += rq[tid + s]; }
        __syncthreads();
    }
    float mean = rs[0] * (1.0f / DIMC);
    float var  = rq[0] * (1.0f / DIMC) - mean * mean;
    float inv  = rsqrtf(var + 1e-5f);
    out[base + c0] = (v0 - mean) * inv * g[c0] + bb[c0];
    out[base + c1] = (v1 - mean) * inv * g[c1] + bb[c1];
}

// ---------------------------------------------------------------------------
extern "C" void kernel_launch(void* const* d_in, const int* in_sizes, int n_in,
                              void* d_out, int out_size, void* d_ws, size_t ws_size,
                              hipStream_t stream)
{
    (void)in_sizes; (void)n_in; (void)out_size; (void)ws_size;

    const float* in_tokens       = (const float*)d_in[0];
    const unsigned char* mask    = (const unsigned char*)d_in[1];
    const float* w_qkv           = (const float*)d_in[2];
    const float* w_out           = (const float*)d_in[3];
    const float* b_out           = (const float*)d_in[4];
    const float* w_narrow        = (const float*)d_in[5];
    const float* b_narrow        = (const float*)d_in[6];
    const float* w_wide          = (const float*)d_in[7];
    const float* b_wide          = (const float*)d_in[8];
    const float* ln1_g           = (const float*)d_in[9];
    const float* ln1_b           = (const float*)d_in[10];
    const float* ff_w            = (const float*)d_in[11];
    const float* ff_b            = (const float*)d_in[12];
    const float* ln2_g           = (const float*)d_in[13];
    const float* ln2_b           = (const float*)d_in[14];
    float* out                   = (float*)d_out;

    // ---- workspace layout (bump alloc, 256B aligned, with safe aliasing)
    char* ws = (char*)d_ws;
    size_t off = 0;
    auto alloc = [&](size_t bytes) -> char* {
        char* p = ws + off;
        off += (bytes + 255) & ~(size_t)255;
        return p;
    };
    __bf16* tokens_bf = (__bf16*)alloc((size_t)ROWS * DIMC * 2);         // reused as ainner
    __bf16* wqkvT     = (__bf16*)alloc((size_t)DIMC * QKVW * 2);         // [1536][512]
    __bf16* woutT     = (__bf16*)alloc((size_t)DIMC * DIMC * 2);
    __bf16* ffwT      = (__bf16*)alloc((size_t)DIMC * DIMC * 2);
    __bf16* wtn       = (__bf16*)alloc((size_t)9 * DIMC * DIMC * 2);     // [t][o][i]
    __bf16* wtw       = (__bf16*)alloc((size_t)9 * DIMC * DIMC * 2);
    float*  qkvF      = (float*) alloc((size_t)ROWS * QKVW * 4);         // reused: nacc/wacc/ffout
    __bf16* qp        = (__bf16*)alloc((size_t)ROWS * DIMC * 2);         // reused as xm
    __bf16* vpT       = (__bf16*)alloc((size_t)ROWS * DIMC * 2);         // reused as ln1b
    __bf16* kT        = (__bf16*)alloc((size_t)ROWS * DIMC * 2);
    __bf16* ctxT      = (__bf16*)alloc((size_t)32 * DHEAD * DHEAD * 2);
    float*  tok       = (float*) alloc((size_t)ROWS * DIMC * 4);
    // aliases (all writes strictly after last read of the aliased buffer)
    __bf16* ainner = tokens_bf;
    float*  nacc   = qkvF;
    float*  wacc   = qkvF + (size_t)ROWS * DIMC;
    float*  ffout  = qkvF + 2 * (size_t)ROWS * DIMC;
    __bf16* xm     = qp;
    __bf16* ln1b   = vpT;

    auto cdiv = [](long long a, long long b) -> unsigned { return (unsigned)((a + b - 1) / b); };

    // ---- precision conversions (weights transposed to B^T form)
    cvt_f32_bf16<<<cdiv((long long)ROWS * DIMC, 256), 256, 0, stream>>>(in_tokens, tokens_bf, (long long)ROWS * DIMC);
    cvt_transpose_f32_bf16<<<cdiv((long long)DIMC * QKVW, 256), 256, 0, stream>>>(w_qkv, wqkvT, DIMC, QKVW);
    cvt_transpose_f32_bf16<<<cdiv((long long)DIMC * DIMC, 256), 256, 0, stream>>>(w_out, woutT, DIMC, DIMC);
    cvt_transpose_f32_bf16<<<cdiv((long long)DIMC * DIMC, 256), 256, 0, stream>>>(ff_w, ffwT, DIMC, DIMC);
    conv_w_transpose<<<cdiv((long long)9 * DIMC * DIMC, 256), 256, 0, stream>>>(w_narrow, wtn, (long long)9 * DIMC * DIMC);
    conv_w_transpose<<<cdiv((long long)9 * DIMC * DIMC, 256), 256, 0, stream>>>(w_wide, wtw, (long long)9 * DIMC * DIMC);

    // ---- qkv = tokens @ w_qkv   [8192 x 1536]
    {
        dim3 g(cdiv(QKVW, BN), cdiv(ROWS, BM), 1);
        gemm_bf16_wmma<<<g, 256, 0, stream>>>(tokens_bf, wqkvT, nullptr, qkvF, nullptr,
            ROWS, QKVW, DIMC, DIMC, DIMC, QKVW,
            0, 0, 1, 0, 0, 0, 0, 0);
    }

    // ---- attention pre-processing (rope, softmaxes, masking)
    attn_pre<<<dim3(SEQ, HEADS, BATCH), 64, 0, stream>>>(qkvF, mask, qp, vpT);
    k_softmax<<<dim3(DHEAD, HEADS, BATCH), 256, 0, stream>>>(qkvF, mask, kT);

    // ---- context[z,d,e] = sum_n kT[z,d,n] * v[z,n,e]  -> stored transposed ctxT[z,e,d]
    {
        dim3 g(cdiv(DHEAD, BN), cdiv(DHEAD, BM), 32);
        gemm_bf16_wmma<<<g, 256, 0, stream>>>(kT, vpT, nullptr, nullptr, ctxT,
            DHEAD, DHEAD, SEQ, SEQ, SEQ, DHEAD,
            (long long)DHEAD * SEQ, (long long)DHEAD * SEQ,
            1, (long long)DHEAD * DHEAD, 0, 0, 0, FLAG_BF16OUT | FLAG_CTRANS);
    }

    // ---- out[z,n,e] = qp[z,n,:] @ ctx[z]  -> ainner[b,n,h*64+e]
    {
        dim3 g(cdiv(DHEAD, BN), cdiv(SEQ, BM), 32);
        gemm_bf16_wmma<<<g, 256, 0, stream>>>(qp, ctxT, nullptr, nullptr, ainner,
            SEQ, DHEAD, DHEAD, DHEAD, DHEAD, DIMC,
            (long long)SEQ * DHEAD, (long long)DHEAD * DHEAD,
            HEADS, (long long)SEQ * DIMC, DHEAD, 0, 0, FLAG_BF16OUT);
    }

    // ---- tok = ainner @ w_out + b_out   [8192 x 512] f32
    dim3 g512(cdiv(DIMC, BN), cdiv(ROWS, BM), 1);
    gemm_bf16_wmma<<<g512, 256, 0, stream>>>(ainner, woutT, b_out, tok, nullptr,
        ROWS, DIMC, DIMC, DIMC, DIMC, DIMC,
        0, 0, 1, 0, 0, 0, 0, FLAG_BIAS);

    // ---- masked conv input
    make_conv_in<<<cdiv((long long)ROWS * DIMC, 256), 256, 0, stream>>>(tok, mask, xm);

    // ---- narrow conv as 9 row-shifted GEMMs accumulated in f32 (pad 4)
    for (int t = 0; t < 9; ++t) {
        gemm_bf16_wmma<<<g512, 256, 0, stream>>>(xm, wtn + (size_t)t * DIMC * DIMC, b_narrow,
            nacc, nullptr, ROWS, DIMC, DIMC, DIMC, DIMC, DIMC,
            0, 0, 1, 0, 0, t - 4, SEQ, (t == 0) ? FLAG_BIAS : FLAG_BETA1);
    }
    // ---- wide conv (dilation 5, effective pad 20)
    for (int t = 0; t < 9; ++t) {
        gemm_bf16_wmma<<<g512, 256, 0, stream>>>(xm, wtw + (size_t)t * DIMC * DIMC, b_wide,
            wacc, nullptr, ROWS, DIMC, DIMC, DIMC, DIMC, DIMC,
            0, 0, 1, 0, 0, t * 5 - 20, SEQ, (t == 0) ? FLAG_BIAS : FLAG_BETA1);
    }

    // ---- residual + gelu(branches) + layernorm1 -> bf16
    combine_ln1<<<ROWS, 256, 0, stream>>>(tok, nacc, wacc, ln1_g, ln1_b, ln1b);

    // ---- ff = ln1 @ ff_w + ff_b  (f32)
    gemm_bf16_wmma<<<g512, 256, 0, stream>>>(ln1b, ffwT, ff_b, ffout, nullptr,
        ROWS, DIMC, DIMC, DIMC, DIMC, DIMC,
        0, 0, 1, 0, 0, 0, 0, FLAG_BIAS);

    // ---- layernorm2(gelu(ff)) -> output
    gelu_ln2<<<ROWS, 256, 0, stream>>>(ffout, ln2_g, ln2_b, out);
}
